// WindowAttention_57561151701641
// MI455X (gfx1250) — matrix operand order repaired and checked by reading, not compiled, not run on hardware
//
#include <hip/hip_runtime.h>
#include <math.h>

#define B_ 256
#define H_ 6
#define N_ 256
#define T_ 64
#define D_ 32
#define C_ 192
#define TOPK_ 32
#define SCALE_ 0.17677669529663687f /* 32^-0.5 */
#define EPS_ 1e-10f

typedef __attribute__((ext_vector_type(16))) _Float16 v16h;
typedef __attribute__((ext_vector_type(8)))  float    v8f;
typedef __attribute__((ext_vector_type(4)))  float    f4_t;
typedef __attribute__((ext_vector_type(4)))  int      i4_t;

// ---------------------------------------------------------------------------
// Kernel 1: per (b,h) attention: scores via WMMA, gather+bias+softmax+PFA,
// top-32 rank selection, AmV + lepe -> x workspace.
// ---------------------------------------------------------------------------
__global__ __launch_bounds__(256)
void attn_kernel(const float* __restrict__ qkvp,
                 const float* __restrict__ pfa_value,
                 const float* __restrict__ rpb,
                 const int*   __restrict__ pfa_index,
                 const int*   __restrict__ rpi,
                 float* __restrict__ xws,
                 float* __restrict__ out_topv,
                 int*   __restrict__ out_idx)
{
    __shared__ __align__(16) _Float16 kh[N_ * D_];   // 16 KB
    __shared__ __align__(16) _Float16 vh[N_ * D_];   // 16 KB
    __shared__ __align__(16) _Float16 qh[16 * D_];   // 1 KB
    __shared__ __align__(16) float    sc[16 * N_];   // 16 KB
    __shared__ float pb[16 * T_];                    // 4 KB
    __shared__ int   ib[16 * T_];                    // 4 KB
    __shared__ float selv[16 * TOPK_];               // 2 KB
    __shared__ int   seli[16 * TOPK_];               // 2 KB

    const int bh   = blockIdx.x;
    const int b    = bh / H_;
    const int h    = bh % H_;
    const int tid  = threadIdx.x;
    const int lane = tid & 31;
    const int wave = tid >> 5;
    const int lm   = lane & 15;
    const int g    = lane >> 4;

    // ---- stage K, V for this (b,h) into LDS as f16; thread tid owns row n=tid
    {
        const int n = tid;
        const size_t base = ((size_t)b * N_ + n) * (4 * C_) + (size_t)h * D_;
        const f4_t* krow = (const f4_t*)(qkvp + base + C_);
        const f4_t* vrow = (const f4_t*)(qkvp + base + 2 * C_);
#pragma unroll
        for (int j = 0; j < 8; ++j) {
            f4_t kf = krow[j];
            f4_t vf = vrow[j];
#pragma unroll
            for (int e = 0; e < 4; ++e) {
                kh[n * D_ + j * 4 + e] = (_Float16)kf[e];
                vh[n * D_ + j * 4 + e] = (_Float16)vf[e];
            }
        }
    }
    __syncthreads();

    for (int tile = 0; tile < N_ / 16; ++tile) {
        // ---- load 16 Q rows (pre-scaled) as f16
#pragma unroll
        for (int rep = 0; rep < 2; ++rep) {
            int idx = tid + rep * 256;           // 512 elements total
            int qr = idx >> 5, qd = idx & 31;
            int row_g = tile * 16 + qr;
            qh[idx] = (_Float16)(qkvp[((size_t)b * N_ + row_g) * (4 * C_) + h * D_ + qd] * SCALE_);
        }
        __syncthreads();

        // ---- WMMA: 16 x 256 score tile; wave w computes column tiles 2w, 2w+1
        {
            union { v16h v; f4_t f[2]; } Au, Bu;
            const f4_t* qrow = (const f4_t*)(qh + lm * D_);
            Au.f[0] = qrow[g];          // halves K = g*8   .. g*8+7
            Au.f[1] = qrow[2 + g];      // halves K = 16+g*8 .. 16+g*8+7
#pragma unroll
            for (int cc = 0; cc < 2; ++cc) {
                const int c = wave * 2 + cc;
                const f4_t* krow2 = (const f4_t*)(kh + (c * 16 + lm) * D_);
                Bu.f[0] = krow2[g];         // halves K = g*8 .. g*8+7
                Bu.f[1] = krow2[2 + g];     // halves K = 16+g*8 .. 16+g*8+7
                v8f acc = {};
                acc = __builtin_amdgcn_wmma_f32_16x16x32_f16(
                        false, Au.v, false, Bu.v, (short)0, acc, false, false);
#pragma unroll
                for (int r = 0; r < 8; ++r)
                    sc[(r + g * 8) * N_ + c * 16 + lm] = acc[r];
            }
        }
        __syncthreads();

        // ---- gather + relative-position bias + softmax + PFA renorm
        const int rr    = tid >> 4;           // row within tile
        const int ss    = tid & 15;           // 16 threads per row, 4 t-slots each
        const int row_g = tile * 16 + rr;
        const size_t poff = ((size_t)(b * H_ + h) * N_ + row_g) * T_ + ss * 4;
        const i4_t idx4 = *(const i4_t*)(pfa_index + poff);
        const f4_t val4 = *(const f4_t*)(pfa_value + poff);

        float a[4];
#pragma unroll
        for (int j = 0; j < 4; ++j) {
            const int idx = idx4[j];
            const float bias = rpb[rpi[row_g * N_ + idx] * H_ + h];
            a[j] = sc[rr * N_ + idx] + bias;
        }
        float mx = fmaxf(fmaxf(a[0], a[1]), fmaxf(a[2], a[3]));
#pragma unroll
        for (int o = 8; o >= 1; o >>= 1) mx = fmaxf(mx, __shfl_xor(mx, o, 16));
        float p[4]; float z = 0.f;
#pragma unroll
        for (int j = 0; j < 4; ++j) { p[j] = __expf(a[j] - mx); z += p[j]; }
#pragma unroll
        for (int o = 8; o >= 1; o >>= 1) z += __shfl_xor(z, o, 16);
        float s2 = 0.f;
#pragma unroll
        for (int j = 0; j < 4; ++j) { p[j] = (p[j] / z) * val4[j]; s2 += p[j]; }
#pragma unroll
        for (int o = 8; o >= 1; o >>= 1) s2 += __shfl_xor(s2, o, 16);
#pragma unroll
        for (int j = 0; j < 4; ++j) {
            p[j] = (p[j] + EPS_) / (s2 + EPS_);
            pb[rr * T_ + ss * 4 + j] = p[j];
            ib[rr * T_ + ss * 4 + j] = idx4[j];
        }
        __syncthreads();

        // ---- top-32 of 64 by stable rank counting (rank == output slot)
#pragma unroll
        for (int j = 0; j < 4; ++j) {
            const int   t  = ss * 4 + j;
            const float pt = p[j];
            int cnt = 0;
            for (int u = 0; u < T_; ++u) {
                const float pu = pb[rr * T_ + u];
                cnt += (pu > pt) || (pu == pt && u < t);
            }
            if (cnt < TOPK_) {
                const size_t ob = ((size_t)(b * H_ + h) * N_ + row_g) * TOPK_ + cnt;
                out_topv[ob] = pt;
                out_idx[ob]  = idx4[j];
                selv[rr * TOPK_ + cnt] = pt;
                seli[rr * TOPK_ + cnt] = idx4[j];
            }
        }
        __syncthreads();

        // ---- AmV (gather V rows) + lepe -> x workspace [B,N,C]
        {
            const int r2 = tid >> 4;
            const int dp = tid & 15;
            float acc0 = 0.f, acc1 = 0.f;
#pragma unroll
            for (int j = 0; j < TOPK_; ++j) {
                const float w = selv[r2 * TOPK_ + j];
                const int   n = seli[r2 * TOPK_ + j];
                acc0 += w * (float)vh[n * D_ + dp];
                acc1 += w * (float)vh[n * D_ + dp + 16];
            }
            const int row2 = tile * 16 + r2;
            const size_t qb = ((size_t)b * N_ + row2) * (4 * C_) + 3 * C_ + h * D_;
            acc0 += qkvp[qb + dp];
            acc1 += qkvp[qb + dp + 16];
            const size_t xb = ((size_t)b * N_ + row2) * C_ + h * D_;
            xws[xb + dp]      = acc0;
            xws[xb + dp + 16] = acc1;
        }
        __syncthreads();
    }
}

// ---------------------------------------------------------------------------
// Kernel 2: out = X @ W^T + b  via WMMA.  Block: 64 rows x 96 cols.
// ---------------------------------------------------------------------------
__global__ __launch_bounds__(256)
void proj_kernel(const float* __restrict__ xws,
                 const float* __restrict__ proj_w,
                 const float* __restrict__ proj_b,
                 float* __restrict__ out)
{
    __shared__ __align__(16) _Float16 ah[64 * C_];   // 24 KB
    __shared__ __align__(16) _Float16 wh[96 * C_];   // 36 KB

    const int tid  = threadIdx.x;
    const int lane = tid & 31, wave = tid >> 5;
    const int lm   = lane & 15, g = lane >> 4;
    const int m0   = blockIdx.x * 64;
    const int n0   = blockIdx.y * 96;

    for (int i = tid; i < 64 * C_; i += 256)
        ah[i] = (_Float16)xws[(size_t)m0 * C_ + i];
    for (int i = tid; i < 96 * C_; i += 256)
        wh[i] = (_Float16)proj_w[(size_t)n0 * C_ + i];
    __syncthreads();

    // 4 x 6 = 24 wave tiles; each wave handles 3
#pragma unroll
    for (int q = 0; q < 3; ++q) {
        const int tf = wave * 3 + q;
        const int rt = tf / 6, ct = tf % 6;
        v8f acc = {};
        const f4_t* arow = (const f4_t*)(ah + (rt * 16 + lm) * C_);
        const f4_t* wrow = (const f4_t*)(wh + (ct * 16 + lm) * C_);
#pragma unroll
        for (int kk = 0; kk < 6; ++kk) {
            union { v16h v; f4_t f[2]; } Au, Bu;
            Au.f[0] = arow[4 * kk + g];
            Au.f[1] = arow[4 * kk + 2 + g];
            Bu.f[0] = wrow[4 * kk + g];
            Bu.f[1] = wrow[4 * kk + 2 + g];
            acc = __builtin_amdgcn_wmma_f32_16x16x32_f16(
                    false, Au.v, false, Bu.v, (short)0, acc, false, false);
        }
        const int ncol = n0 + ct * 16 + lm;
        const float bias = proj_b[ncol];
#pragma unroll
        for (int r = 0; r < 8; ++r) {
            const int mrow = m0 + rt * 16 + g * 8 + r;
            out[(size_t)mrow * C_ + ncol] = acc[r] + bias;
        }
    }
}

// ---------------------------------------------------------------------------
extern "C" void kernel_launch(void* const* d_in, const int* in_sizes, int n_in,
                              void* d_out, int out_size, void* d_ws, size_t ws_size,
                              hipStream_t stream)
{
    const float* qkvp      = (const float*)d_in[0];
    const float* pfa_value = (const float*)d_in[1];
    const float* rpb       = (const float*)d_in[2];
    const float* proj_w    = (const float*)d_in[3];
    const float* proj_b    = (const float*)d_in[4];
    const int*   pfa_index = (const int*)d_in[5];
    const int*   rpi       = (const int*)d_in[6];

    float* out      = (float*)d_out;
    float* out_x    = out;                                            // [B,N,C]
    float* out_topv = out + (size_t)B_ * N_ * C_;                     // [B,H,N,32]
    int*   out_idx  = (int*)(out_topv + (size_t)B_ * H_ * N_ * TOPK_);// [B,H,N,32]

    float* xws = (float*)d_ws;  // [B,N,C] pre-projection x (50.3 MB)

    attn_kernel<<<dim3(B_ * H_), 256, 0, stream>>>(
        qkvp, pfa_value, rpb, pfa_index, rpi, xws, out_topv, out_idx);

    proj_kernel<<<dim3((B_ * N_) / 64, 2), 256, 0, stream>>>(
        xws, proj_w, proj_b, out_x);
}
